// SelfAttention_3977139716573
// MI455X (gfx1250) — matrix operand order repaired and checked by reading, not compiled, not run on hardware
//
#include <hip/hip_runtime.h>
#include <cstdint>
#include <cstddef>

// Problem constants (match reference)
#define BATCH 4
#define SEQ   2048
#define EMB   1024
#define NHEAD 16
#define DH    64
#define BS    (BATCH * SEQ)   // 8192 rows
#define E3    (3 * EMB)       // 3072

typedef __attribute__((ext_vector_type(16))) __bf16       v16bf;
typedef __attribute__((ext_vector_type(8)))  float        v8f;
typedef __attribute__((ext_vector_type(8)))  unsigned int u32x8;
typedef __attribute__((ext_vector_type(4)))  unsigned int u32x4;
typedef __attribute__((ext_vector_type(8)))  int          i32x8;
typedef __attribute__((ext_vector_type(4)))  int          i32x4;

#if defined(__has_builtin)
#if __has_builtin(__builtin_amdgcn_tensor_load_to_lds)
#define HAVE_TDM 1
#endif
#endif
#ifndef HAVE_TDM
#define HAVE_TDM 0
#endif

// ---------------------------------------------------------------------------
// helpers
// ---------------------------------------------------------------------------
__device__ __forceinline__ unsigned short f32_to_bf16(float f) {
  unsigned int u = __builtin_bit_cast(unsigned int, f);
  u += 0x7fffu + ((u >> 16) & 1u);          // round-to-nearest-even
  return (unsigned short)(u >> 16);
}

__device__ __forceinline__ v8f zero8() {
  v8f z;
#pragma unroll
  for (int i = 0; i < 8; ++i) z[i] = 0.0f;
  return z;
}

__device__ __forceinline__ v8f wmma_bf16(v16bf a, v16bf b, v8f c) {
  // D = A(16x32 bf16) * B(32x16 bf16) + C(16x16 f32)
  return __builtin_amdgcn_wmma_f32_16x16x32_bf16(
      /*neg_a=*/false, a, /*neg_b=*/false, b,
      /*c_mod=*/(short)0, c, /*reuse_a=*/false, /*reuse_b=*/false);
}

// A fragment: 16(M) x 32(K) bf16, source row-major with given element stride.
// ISA layout: lane<16 holds row m=lane, K {0..7,16..23}; lane>=16 row m=lane-16,
// K {8..15,24..31}; two bf16 per VGPR. Contiguous 4xdword runs -> b128 loads.
__device__ __forceinline__ v16bf load_a_frag(const unsigned short* base, int stride) {
  const int lane = threadIdx.x & 31;
  const int r = lane & 15, h2 = lane >> 4;
  const unsigned short* rowp = base + (size_t)r * stride;
  u32x8 u;
#pragma unroll
  for (int j = 0; j < 8; ++j) {
    const int kb = (j < 4) ? (8 * h2 + 2 * j) : (16 + 8 * h2 + 2 * (j - 4));
    u[j] = *(const unsigned int*)(rowp + kb);
  }
  return __builtin_bit_cast(v16bf, u);
}

// B fragment: 32(K) x 16(N) with B[k][n] = W[n][k], W row-major [N][K].
// ISA layout: lane<16 holds column n=lane, K {0..7,16..23}; lane>=16 column
// n=lane-16, K {8..15,24..31} (2/VGPR), matching load_a_frag.
__device__ __forceinline__ v16bf load_b_frag(const unsigned short* base, int stride) {
  const int lane = threadIdx.x & 31;
  const int n = lane & 15, h2 = lane >> 4;
  const unsigned short* rowp = base + (size_t)n * stride;
  u32x8 u;
#pragma unroll
  for (int t = 0; t < 8; ++t) {
    const int kb = (t < 4) ? (8 * h2 + 2 * t) : (16 + 8 * h2 + 2 * (t - 4));
    u[t] = *(const unsigned int*)(rowp + kb);
  }
  return __builtin_bit_cast(v16bf, u);
}

// ---------------------------------------------------------------------------
// Tensor Data Mover: 2D tile DMA (global -> LDS), bf16 elements.
// D# bitfields per CDNA5 ISA ch.8: group0 {count=1, lds_addr, global_addr,
// type=2}; group1 {data_size=2B, tensor_dim0/1, tile_dim0/1, dim0_stride}.
// All dims/strides in elements (data_size units).
// ---------------------------------------------------------------------------
#if HAVE_TDM
__device__ __forceinline__ void tdm_load_2d(unsigned lds_off, const void* gptr,
                                            unsigned tensor_d0, unsigned tensor_d1,
                                            unsigned long long stride0,
                                            unsigned tile_d0, unsigned tile_d1) {
  const unsigned long long ga = (unsigned long long)gptr;
  u32x4 g0;
  g0[0] = 1u;                                              // count=1, user mode
  g0[1] = lds_off;                                         // lds_addr [63:32]
  g0[2] = (unsigned)ga;                                    // global_addr lo
  g0[3] = ((unsigned)(ga >> 32) & 0x01ffffffu) | 0x80000000u;  // ga[56:32] | type=2

  u32x8 g1u;
  g1u[0] = (1u << 16);                                     // data_size=1 (2 bytes)
  g1u[1] = (tensor_d0 & 0xffffu) << 16;                    // tensor_dim0[15:0]
  g1u[2] = (tensor_d0 >> 16) | ((tensor_d1 & 0xffffu) << 16);
  g1u[3] = (tensor_d1 >> 16) | ((tile_d0 & 0xffffu) << 16);// tile_dim0
  g1u[4] = (tile_d1 & 0xffffu);                            // tile_dim1 (tile_dim2=0)
  g1u[5] = (unsigned)stride0;                              // dim0_stride lo
  g1u[6] = (unsigned)(stride0 >> 32) & 0xffffu;            // dim0_stride hi
  g1u[7] = 0u;                                             // dim1_stride unused (2D)
  const i32x8 g1 = __builtin_bit_cast(i32x8, g1u);
  const i32x4 z4 = {0, 0, 0, 0};
#if defined(__clang_major__) && (__clang_major__ >= 23)
  const i32x8 z8 = {0, 0, 0, 0, 0, 0, 0, 0};
  __builtin_amdgcn_tensor_load_to_lds(g0, g1, z4, z4, z8, 0);
#else
  __builtin_amdgcn_tensor_load_to_lds(g0, g1, z4, z4, 0);
#endif
}
#endif

// Stage one 32-wide KV chunk into LDS. TDM path: wave 0 issues two tensor DMAs.
// Fallback: cooperative vector copies by all 128 threads.
__device__ __forceinline__ void stage_kv(unsigned short* kd, unsigned short* vd,
                                         const unsigned short* ks,
                                         const unsigned short* vs) {
#if HAVE_TDM
  if ((threadIdx.x >> 5) == 0) {
    // K tile: tensor [SEQ][DH] row-major, tile 32 rows x 64 cols (contiguous rows)
    tdm_load_2d((unsigned)(size_t)kd, ks, /*d0=*/DH, /*d1=*/SEQ,
                /*stride0=*/DH, /*tile0=*/DH, /*tile1=*/32);
    // V^T tile: tensor [DH][SEQ] row-major, tile 32 cols x 64 rows -> LDS [64][32]
    tdm_load_2d((unsigned)(size_t)vd, vs, /*d0=*/SEQ, /*d1=*/DH,
                /*stride0=*/SEQ, /*tile0=*/32, /*tile1=*/DH);
  }
#else
  const int t = threadIdx.x;
  {  // K tile is one contiguous 4KB block (32 rows * 64 cols * 2B)
    const u32x4* src = (const u32x4*)ks;
    u32x4* dst = (u32x4*)kd;
    dst[t] = src[t];
    dst[t + 128] = src[t + 128];
  }
  {  // V^T tile: 64 rows of 32 elements, source stride SEQ
    const int row = t >> 1, half = t & 1;
    const u32x4* src = (const u32x4*)(vs + (size_t)row * SEQ + half * 16);
    u32x4* dst = (u32x4*)(vd + row * 32 + half * 16);
    dst[0] = src[0];
    dst[1] = src[1];
  }
#endif
}

__device__ __forceinline__ void stage_wait() {
#if HAVE_TDM
  if ((threadIdx.x >> 5) == 0) __builtin_amdgcn_s_wait_tensorcnt(0);
#endif
}

// ---------------------------------------------------------------------------
// Pass 0: fp32 -> bf16 conversion
// ---------------------------------------------------------------------------
__global__ void cvt_bf16_kernel(const float* __restrict__ src,
                                unsigned short* __restrict__ dst, int n) {
  const int i = blockIdx.x * blockDim.x + threadIdx.x;
  if (i < n) dst[i] = f32_to_bf16(src[i]);
}

// ---------------------------------------------------------------------------
// Pass 1: QKV projection  C[BS,3E] = Xbf @ Winbf^T + b_in
// Each wave: one 16(M) x 64(N) tile, K-loop over EMB in steps of 32.
// Epilogue scatters Q,K as [B,H,S,64] bf16 and V transposed as [B,H,64,S] bf16.
// ---------------------------------------------------------------------------
__global__ __launch_bounds__(128) void qkv_gemm_kernel(
    const unsigned short* __restrict__ xbf,
    const unsigned short* __restrict__ wbf,   // [3E][E] row-major (= w_in layout)
    const float* __restrict__ bias,           // [3E]
    unsigned short* __restrict__ qb,
    unsigned short* __restrict__ kbuf,
    unsigned short* __restrict__ vtb) {
  const int wid = blockIdx.x * 4 + (threadIdx.x >> 5);
  const int NG = E3 / 64;                       // 48 column groups
  const int m0 = (wid / NG) * 16;
  const int n0 = (wid % NG) * 64;
  const int lane = threadIdx.x & 31;
  const int nlo = lane & 15, h2 = lane >> 4;

  v8f acc[4];
#pragma unroll
  for (int nt = 0; nt < 4; ++nt) acc[nt] = zero8();

  for (int kk = 0; kk < EMB; kk += 32) {
    const v16bf a = load_a_frag(xbf + (size_t)m0 * EMB + kk, EMB);
#pragma unroll
    for (int nt = 0; nt < 4; ++nt) {
      const v16bf w = load_b_frag(wbf + (size_t)(n0 + nt * 16) * EMB + kk, EMB);
      acc[nt] = wmma_bf16(a, w, acc[nt]);
    }
  }

#pragma unroll
  for (int nt = 0; nt < 4; ++nt) {
#pragma unroll
    for (int r = 0; r < 8; ++r) {
      const int row = m0 + r + 8 * h2;          // [0, BS)
      const int col = n0 + nt * 16 + nlo;       // [0, 3E)
      const float v = acc[nt][r] + bias[col];
      const int b = row / SEQ, s = row % SEQ;
      const int which = col / EMB;              // uniform per wave
      const int e = col % EMB;
      const int h = e / DH, d = e % DH;
      const int bh = b * NHEAD + h;
      const unsigned short bv = f32_to_bf16(v);
      if (which == 0)      qb[((size_t)bh * SEQ + s) * DH + d]   = bv;
      else if (which == 1) kbuf[((size_t)bh * SEQ + s) * DH + d] = bv;
      else                 vtb[((size_t)bh * DH + d) * SEQ + s]  = bv;   // V^T
    }
  }
}

// ---------------------------------------------------------------------------
// Pass 2: flash attention. Workgroup = 4 waves = 64 query rows of one (b,h).
// KV streamed in 32-wide chunks through double-buffered LDS filled by the
// Tensor Data Mover (wave 0 issues DMA, s_wait_tensorcnt + barrier sync),
// overlapping DMA of chunk j+1 with WMMA compute on chunk j.
// ---------------------------------------------------------------------------
__global__ __launch_bounds__(128) void flash_attn_kernel(
    const unsigned short* __restrict__ qb,    // [B,H,S,64] bf16
    const unsigned short* __restrict__ kbuf,  // [B,H,S,64] bf16
    const unsigned short* __restrict__ vtb,   // [B,H,64,S] bf16
    unsigned short* __restrict__ attn) {      // [B,S,E] bf16
  __shared__ __align__(16) unsigned short s_k[2][32 * DH];  // 2 x 4 KB
  __shared__ __align__(16) unsigned short s_v[2][DH * 32];  // 2 x 4 KB
  __shared__ __align__(16) unsigned short s_p[4][16 * 32];  // 4 x 1 KB

  const int wave = threadIdx.x >> 5;
  const int lane = threadIdx.x & 31;
  const int nlo = lane & 15, h2 = lane >> 4;

  const int grp = blockIdx.x;                   // [0, B*H*(S/64))
  const int bh = grp >> 5;                      // S/64 = 32 row-groups per (b,h)
  const int m0 = ((grp & 31) * 4 + wave) * 16;  // this wave's 16 query rows
  const int b = bh / NHEAD, h = bh % NHEAD;

  const unsigned short* qbase = qb   + (size_t)bh * SEQ * DH;
  const unsigned short* kbase = kbuf + (size_t)bh * SEQ * DH;
  const unsigned short* vbase = vtb  + (size_t)bh * DH * SEQ;

  const v16bf q0 = load_a_frag(qbase + (size_t)m0 * DH, DH);       // d 0..31
  const v16bf q1 = load_a_frag(qbase + (size_t)m0 * DH + 32, DH);  // d 32..63

  v8f acc[4];
#pragma unroll
  for (int nt = 0; nt < 4; ++nt) acc[nt] = zero8();
  float mrow[8], lrow[8];
#pragma unroll
  for (int r = 0; r < 8; ++r) { mrow[r] = -1e30f; lrow[r] = 0.0f; }

  unsigned short* myp = s_p[wave];
  const float scale = 0.125f;                   // 1/sqrt(Dh)
  const int NC = SEQ / 32;                      // 64 chunks

  // prologue: fill buffer 0
  stage_kv(s_k[0], s_v[0], kbase, vbase);
  stage_wait();
  __syncthreads();

  int c = 0;
  for (int chunk = 0; chunk < NC; ++chunk) {
    // prefetch next chunk into the other buffer while we compute on this one
    if (chunk + 1 < NC) {
      const int jn = (chunk + 1) * 32;
      stage_kv(s_k[c ^ 1], s_v[c ^ 1], kbase + (size_t)jn * DH, vbase + jn);
    }
    const unsigned short* kt = s_k[c];
    const unsigned short* vt = s_v[c];

    // scores: S[16 x 32] = Q(16x64) @ K^T(64x32), two 16x16 C fragments
    v8f s0 = zero8(), s1 = zero8();
    s0 = wmma_bf16(q0, load_b_frag(kt, DH), s0);
    s0 = wmma_bf16(q1, load_b_frag(kt + 32, DH), s0);
    s1 = wmma_bf16(q0, load_b_frag(kt + 16 * DH, DH), s1);
    s1 = wmma_bf16(q1, load_b_frag(kt + 16 * DH + 32, DH), s1);

    float p0[8], p1[8];
#pragma unroll
    for (int r = 0; r < 8; ++r) {
      const float a0 = s0[r] * scale, a1 = s1[r] * scale;
      // row max across the 16 lanes holding this row's columns
      float mx = fmaxf(a0, a1);
      mx = fmaxf(mx, __shfl_xor(mx, 1, 16));
      mx = fmaxf(mx, __shfl_xor(mx, 2, 16));
      mx = fmaxf(mx, __shfl_xor(mx, 4, 16));
      mx = fmaxf(mx, __shfl_xor(mx, 8, 16));
      const float mnew = fmaxf(mrow[r], mx);
      const float alpha = __expf(mrow[r] - mnew);
      const float e0 = __expf(a0 - mnew);
      const float e1 = __expf(a1 - mnew);
      float rs = e0 + e1;
      rs += __shfl_xor(rs, 1, 16);
      rs += __shfl_xor(rs, 2, 16);
      rs += __shfl_xor(rs, 4, 16);
      rs += __shfl_xor(rs, 8, 16);
      lrow[r] = lrow[r] * alpha + rs;
      mrow[r] = mnew;
      p0[r] = e0; p1[r] = e1;
#pragma unroll
      for (int nt = 0; nt < 4; ++nt) acc[nt][r] *= alpha;
    }

    // C-layout -> A-layout reshape through this wave's private LDS slice
#pragma unroll
    for (int r = 0; r < 8; ++r) {
      const int m = r + 8 * h2;
      myp[m * 32 + nlo]      = f32_to_bf16(p0[r]);
      myp[m * 32 + 16 + nlo] = f32_to_bf16(p1[r]);
    }
    const v16bf pf = load_a_frag(myp, 32);      // P as 16x32 A fragment

    // out += P(16x32) @ V(32x64): B fragments from LDS V^T tile [64][32]
#pragma unroll
    for (int nt = 0; nt < 4; ++nt) {
      const v16bf vf = load_b_frag(vt + (nt * 16) * 32, 32);
      acc[nt] = wmma_bf16(pf, vf, acc[nt]);
    }

    stage_wait();            // wave 0: DMA for next chunk has landed
    __syncthreads();         // everyone done with buf[c]; buf[c^1] ready
    c ^= 1;
  }

  // normalize and write [B,S,E] bf16 (head-interleaved)
#pragma unroll
  for (int nt = 0; nt < 4; ++nt) {
#pragma unroll
    for (int r = 0; r < 8; ++r) {
      const int m = r + 8 * h2;
      const int s = m0 + m;
      const int col = h * DH + nt * 16 + nlo;
      const float v = acc[nt][r] / lrow[r];
      attn[((size_t)b * SEQ + s) * EMB + col] = f32_to_bf16(v);
    }
  }
}

// ---------------------------------------------------------------------------
// Pass 3: output projection  out[BS,E] = Abf @ Woutbf^T + b_out  (fp32 out)
// ---------------------------------------------------------------------------
__global__ __launch_bounds__(128) void out_proj_kernel(
    const unsigned short* __restrict__ abf,
    const unsigned short* __restrict__ wbf,   // [E][E] row-major (= w_out layout)
    const float* __restrict__ bias,
    float* __restrict__ out) {
  const int wid = blockIdx.x * 4 + (threadIdx.x >> 5);
  const int NG = EMB / 64;                    // 16
  const int m0 = (wid / NG) * 16;
  const int n0 = (wid % NG) * 64;
  const int lane = threadIdx.x & 31;
  const int nlo = lane & 15, h2 = lane >> 4;

  v8f acc[4];
#pragma unroll
  for (int nt = 0; nt < 4; ++nt) acc[nt] = zero8();

  for (int kk = 0; kk < EMB; kk += 32) {
    const v16bf a = load_a_frag(abf + (size_t)m0 * EMB + kk, EMB);
#pragma unroll
    for (int nt = 0; nt < 4; ++nt) {
      const v16bf w = load_b_frag(wbf + (size_t)(n0 + nt * 16) * EMB + kk, EMB);
      acc[nt] = wmma_bf16(a, w, acc[nt]);
    }
  }

#pragma unroll
  for (int nt = 0; nt < 4; ++nt) {
#pragma unroll
    for (int r = 0; r < 8; ++r) {
      const int row = m0 + r + 8 * h2;
      const int col = n0 + nt * 16 + nlo;
      out[(size_t)row * EMB + col] = acc[nt][r] + bias[col];
    }
  }
}

// ---------------------------------------------------------------------------
// launcher
// ---------------------------------------------------------------------------
extern "C" void kernel_launch(void* const* d_in, const int* in_sizes, int n_in,
                              void* d_out, int out_size, void* d_ws, size_t ws_size,
                              hipStream_t stream) {
  (void)in_sizes; (void)n_in; (void)out_size; (void)ws_size;
  const float* x     = (const float*)d_in[0];
  const float* w_in  = (const float*)d_in[1];
  const float* b_in  = (const float*)d_in[2];
  const float* w_out = (const float*)d_in[3];
  const float* b_out = (const float*)d_in[4];
  float* out = (float*)d_out;

  char* ws = (char*)d_ws;
  size_t off = 0;
  auto alloc = [&](size_t bytes) -> void* {
    void* p = ws + off;
    off = (off + bytes + 255) & ~(size_t)255;
    return p;
  };
  unsigned short* x_bf    = (unsigned short*)alloc((size_t)BS * EMB * 2);   // 16 MB
  unsigned short* win_bf  = (unsigned short*)alloc((size_t)E3 * EMB * 2);   //  6 MB
  unsigned short* wout_bf = (unsigned short*)alloc((size_t)EMB * EMB * 2);  //  2 MB
  unsigned short* q_bf    = (unsigned short*)alloc((size_t)BS * EMB * 2);   // 16 MB
  unsigned short* k_bf    = (unsigned short*)alloc((size_t)BS * EMB * 2);   // 16 MB
  unsigned short* vt_bf   = (unsigned short*)alloc((size_t)BS * EMB * 2);   // 16 MB
  unsigned short* attn_bf = (unsigned short*)alloc((size_t)BS * EMB * 2);   // 16 MB

  // Pass 0: conversions (sizes all divide 256)
  cvt_bf16_kernel<<<(BS * EMB) / 256, 256, 0, stream>>>(x, x_bf, BS * EMB);
  cvt_bf16_kernel<<<(E3 * EMB) / 256, 256, 0, stream>>>(w_in, win_bf, E3 * EMB);
  cvt_bf16_kernel<<<(EMB * EMB) / 256, 256, 0, stream>>>(w_out, wout_bf, EMB * EMB);

  // Pass 1: QKV projection — (BS/16)*(3E/64) = 24576 waves, 4 waves/block
  const int qkv_waves = (BS / 16) * (E3 / 64);
  qkv_gemm_kernel<<<qkv_waves / 4, 128, 0, stream>>>(x_bf, win_bf, b_in,
                                                     q_bf, k_bf, vt_bf);

  // Pass 2: flash attention — B*H*(S/64) = 2048 workgroups of 4 waves
  const int attn_groups = BATCH * NHEAD * (SEQ / 64);
  flash_attn_kernel<<<attn_groups, 128, 0, stream>>>(q_bf, k_bf, vt_bf, attn_bf);

  // Pass 3: output projection — (BS/16)*(E/64) = 8192 waves
  const int op_waves = (BS / 16) * (EMB / 64);
  out_proj_kernel<<<op_waves / 4, 128, 0, stream>>>(attn_bf, wout_bf, b_out, out);
}
